// SDEembedding_53936199303291
// MI455X (gfx1250) — compile-verified, matrix operand order checked by reading, not run on hardware
//
#include <hip/hip_runtime.h>
#include <hip/hip_bf16.h>

typedef _Float16 half_t;
typedef __attribute__((ext_vector_type(16))) _Float16 v16h;
typedef __attribute__((ext_vector_type(8)))  _Float16 v8h;
typedef __attribute__((ext_vector_type(8)))  float    v8f;

#define B_  8
#define L_  512
#define C_  4000
#define D_  512
#define V_  10000
#define M_  (B_ * L_)      // 4096
#define VP_ 10048          // V padded to multiple of 64 (ctx K-tile)

union V16H { v16h v; v8h h[2]; };

// Load a 16x32 f16 WMMA fragment (A or B) where the 16 "row" dim maps to
// lanes 0..15 and K is contiguous in memory (NT layout).
// ISA layout (16-bit A 16x32): lanes 0-15 hold K = k..k+7 (elts 0-7) and
// K = k+16..k+23 (elts 8-15); lanes 16-31 hold K+8 and K+24.
__device__ inline v16h load_frag_nt(const half_t* __restrict__ base,
                                    int row_stride, int row, int k, int lane) {
  const half_t* p = base + (size_t)row * (size_t)row_stride + k + ((lane >> 4) * 8);
  V16H r;
  r.h[0] = *(const v8h*)(p);
  r.h[1] = *(const v8h*)(p + 16);
  return r.v;
}

__device__ inline v16h load_frag_lds(const half_t* base, int row_stride, int k, int lane) {
  const half_t* p = base + (size_t)(lane & 15) * row_stride + k + ((lane >> 4) * 8);
  V16H r;
  r.h[0] = *(const v8h*)(p);
  r.h[1] = *(const v8h*)(p + 16);
  return r.v;
}

__device__ inline v8f wmma_f16(v16h a, v16h b, v8f c) {
  return __builtin_amdgcn_wmma_f32_16x16x32_f16(false, a, false, b, (short)0, c, false, false);
}

// ---------------------------------------------------------------------------
// 1) f32 -> f16 elementwise convert
// ---------------------------------------------------------------------------
__global__ __launch_bounds__(256)
void convert_f2h(const float* __restrict__ in, half_t* __restrict__ out, int n) {
  int i = blockIdx.x * 256 + threadIdx.x;
  int stride = gridDim.x * 256;
  for (; i < n; i += stride) out[i] = (half_t)in[i];
}

// ---------------------------------------------------------------------------
// 2) E[V,D] f32 -> E^T[D,VP] f16 (tiled transpose, zero-pad v in [V, VP))
// ---------------------------------------------------------------------------
__global__ __launch_bounds__(256)
void transpose_emb(const float* __restrict__ emb, half_t* __restrict__ ehT) {
  __shared__ half_t tile[32][33];
  int vb = blockIdx.x * 32;   // v base (VP/32 = 314 blocks)
  int db = blockIdx.y * 32;   // d base (D/32 = 16 blocks)
  int tx = threadIdx.x & 31, ty = threadIdx.x >> 5;   // 32 x 8
  #pragma unroll
  for (int i = 0; i < 32; i += 8) {
    int v = vb + ty + i, d = db + tx;
    tile[ty + i][tx] = (v < V_) ? (half_t)emb[(size_t)v * D_ + d] : (half_t)0.0f;
  }
  __syncthreads();
  #pragma unroll
  for (int i = 0; i < 32; i += 8) {
    int d = db + ty + i, v = vb + tx;
    ehT[(size_t)d * VP_ + v] = tile[tx][ty + i];
  }
}

// ---------------------------------------------------------------------------
// 3) q = tanh(x[M,C] @ W^T[C,D]); write q f32 (residual) and q f16 (stage 2)
//    One wave per 16x16 output tile; grid = (M/16)*(D/16) = 8192.
// ---------------------------------------------------------------------------
__global__ __launch_bounds__(32)
void proj_gemm(const half_t* __restrict__ xh, const half_t* __restrict__ wh,
               float* __restrict__ qf, half_t* __restrict__ qh) {
  int tile = blockIdx.x;
  int m0 = (tile % (M_ / 16)) * 16;
  int n0 = (tile / (M_ / 16)) * 16;
  int lane = threadIdx.x;
  int col = lane & 15, hl = lane >> 4;

  v8f acc = {};
  for (int k = 0; k < C_; k += 32) {
    v16h a = load_frag_nt(xh, C_, m0 + col, k, lane);   // x rows, K=c contiguous
    v16h b = load_frag_nt(wh, C_, n0 + col, k, lane);   // W rows (N=d), K=c contiguous
    acc = wmma_f16(a, b, acc);
  }
  #pragma unroll
  for (int j = 0; j < 8; ++j) {
    int m = m0 + j + 8 * hl;                 // C-frag: VGPR j -> row j (+8 for hi lanes)
    float t = tanhf(acc[j]);
    size_t idx = (size_t)m * D_ + n0 + col;
    qf[idx] = t;
    qh[idx] = (half_t)t;
  }
}

// ---------------------------------------------------------------------------
// 4) Fused attention: out = softmax(q E^T) E + q
//    One 16-row M-tile per block; 4 cooperating waves. V tile = 64:
//    wave w computes S for v-cols [v0+16w, v0+16w+16) (no duplicate work),
//    softmax stats merged via LDS, P staged in LDS, each wave accumulates
//    its own 128-wide D slice (8 acc fragments, K=64 per V tile).
// ---------------------------------------------------------------------------
__global__ __launch_bounds__(128)
void attn_fused(const half_t* __restrict__ qh, const half_t* __restrict__ eh,
                const half_t* __restrict__ ehT, const float* __restrict__ qf,
                float* __restrict__ out) {
  __shared__ alignas(16) half_t qtile[16][D_ + 8];  // stride 520 halves: conflict-free
  __shared__ alignas(16) half_t ptile[16][72];      // stride 72 halves:  conflict-free
  __shared__ float smaxLds[4][16];
  __shared__ float psumLds[4][16];

  int wave = threadIdx.x >> 5;       // 0..3
  int lane = threadIdx.x & 31;
  int col = lane & 15, hl = lane >> 4;
  int m0 = blockIdx.x * 16;

  // Cooperative load of the 16x512 q tile (4 rows per wave).
  for (int r = 0; r < 4; ++r) {
    int row = wave * 4 + r;
    const v8h* src = (const v8h*)(qh + (size_t)(m0 + row) * D_);
    v8h* dst = (v8h*)(&qtile[row][0]);
    dst[lane]      = src[lane];        // 64 v8h chunks per row, 2 per lane
    dst[lane + 32] = src[lane + 32];
  }
  __syncthreads();

  float rowmax[8], rowsum[8];
  #pragma unroll
  for (int j = 0; j < 8; ++j) { rowmax[j] = -3.0e38f; rowsum[j] = 0.0f; }
  v8f acc[8];
  #pragma unroll
  for (int t = 0; t < 8; ++t) acc[t] = (v8f){};

  for (int v0 = 0; v0 < V_; v0 += 64) {
    // ---- Phase A: this wave's 16x16 logit tile, columns v0+16*wave+col
    v8f s = {};
    int vc = v0 + 16 * wave + col;
    int er = (vc < V_) ? vc : (V_ - 1);
    for (int k = 0; k < D_; k += 32) {
      v16h a = load_frag_lds(&qtile[0][0], D_ + 8, k, lane);
      v16h b = load_frag_nt(eh, D_, er, k, lane);   // E rows, K=d contiguous
      s = wmma_f16(a, b, s);
    }
    const bool oob = vc >= V_;
    float tmax[8];
    #pragma unroll
    for (int j = 0; j < 8; ++j) {
      if (oob) s[j] = -3.0e38f;
      float t = s[j];
      #pragma unroll
      for (int o = 1; o < 16; o <<= 1) t = fmaxf(t, __shfl_xor(t, o, 16));
      tmax[j] = t;
      if (col == 0) smaxLds[wave][j + 8 * hl] = t;   // lanes 0 & 16 cover 16 rows
    }
    __syncthreads();   // barrier 1: tile maxima visible

    // ---- Phase B: merge maxima, exponentiate, stage P, partial sums
    float c[8];
    #pragma unroll
    for (int j = 0; j < 8; ++j) {
      int row = j + 8 * hl;
      float g = fmaxf(fmaxf(smaxLds[0][row], smaxLds[1][row]),
                      fmaxf(smaxLds[2][row], smaxLds[3][row]));
      float mnew = fmaxf(rowmax[j], g);
      c[j] = __expf(rowmax[j] - mnew);
      rowmax[j] = mnew;
      float p = __expf(s[j] - mnew);
      float ps = p;
      #pragma unroll
      for (int o = 1; o < 16; o <<= 1) ps += __shfl_xor(ps, o, 16);
      if (col == 0) psumLds[wave][row] = ps;
      ptile[row][16 * wave + col] = (half_t)p;       // C-layout -> A-layout staging
    }
    __syncthreads();   // barrier 2: P + partial sums visible

    // ---- Phase C: combine sums, rescale, ctx += P(16x64) @ E^T slice
    #pragma unroll
    for (int j = 0; j < 8; ++j) {
      int row = j + 8 * hl;
      rowsum[j] = rowsum[j] * c[j] + (psumLds[0][row] + psumLds[1][row]) +
                                     (psumLds[2][row] + psumLds[3][row]);
    }
    #pragma unroll
    for (int t = 0; t < 8; ++t) {
      #pragma unroll
      for (int j = 0; j < 8; ++j) acc[t][j] *= c[j];
    }
    v16h pf0 = load_frag_lds(&ptile[0][0], 72, 0,  lane);
    v16h pf1 = load_frag_lds(&ptile[0][0], 72, 32, lane);
    #pragma unroll
    for (int nt = 0; nt < 8; ++nt) {
      int drow = (wave * 8 + nt) * 16 + col;         // N index = d (this wave's D slice)
      v16h bf0 = load_frag_nt(ehT, VP_, drow, v0,      lane);
      v16h bf1 = load_frag_nt(ehT, VP_, drow, v0 + 32, lane);
      acc[nt] = wmma_f16(pf0, bf0, acc[nt]);
      acc[nt] = wmma_f16(pf1, bf1, acc[nt]);
    }
  }

  // ---- epilogue: normalize + residual (this wave's 128 D columns)
  float inv[8];
  #pragma unroll
  for (int j = 0; j < 8; ++j) inv[j] = 1.0f / rowsum[j];
  #pragma unroll
  for (int nt = 0; nt < 8; ++nt) {
    #pragma unroll
    for (int j = 0; j < 8; ++j) {
      size_t idx = (size_t)(m0 + j + 8 * hl) * D_ + (wave * 8 + nt) * 16 + col;
      out[idx] = acc[nt][j] * inv[j] + qf[idx];
    }
  }
}

// ---------------------------------------------------------------------------
extern "C" void kernel_launch(void* const* d_in, const int* in_sizes, int n_in,
                              void* d_out, int out_size, void* d_ws, size_t ws_size,
                              hipStream_t stream) {
  const float* x = (const float*)d_in[0];   // [B,L,C]
  const float* W = (const float*)d_in[1];   // [D,C]
  const float* E = (const float*)d_in[2];   // [V,D]
  float* out = (float*)d_out;               // [B,L,D]

  char* ws = (char*)d_ws;
  size_t off = 0;
  auto take = [&](size_t bytes) -> char* {
    char* p = ws + off;
    off = (off + bytes + 255) & ~(size_t)255;
    return p;
  };
  const int NX = M_ * C_;        // 16,384,000
  const int NW = D_ * C_;        //  2,048,000
  const int NE = V_ * D_;        //  5,120,000
  half_t* xh  = (half_t*)take((size_t)NX * 2);
  half_t* wh  = (half_t*)take((size_t)NW * 2);
  half_t* eh  = (half_t*)take((size_t)NE * 2);
  half_t* ehT = (half_t*)take((size_t)D_ * VP_ * 2);
  float*  qf  = (float*) take((size_t)M_ * D_ * 4);
  half_t* qh  = (half_t*)take((size_t)M_ * D_ * 2);

  convert_f2h<<<(NX + 255) / 256, 256, 0, stream>>>(x, xh, NX);
  convert_f2h<<<(NW + 255) / 256, 256, 0, stream>>>(W, wh, NW);
  convert_f2h<<<(NE + 255) / 256, 256, 0, stream>>>(E, eh, NE);
  transpose_emb<<<dim3(VP_ / 32, D_ / 32), 256, 0, stream>>>(E, ehT);
  proj_gemm<<<(M_ / 16) * (D_ / 16), 32, 0, stream>>>(xh, wh, qf, qh);
  attn_fused<<<M_ / 16, 128, 0, stream>>>(qh, eh, ehT, qf, out);
}
